// Head_72748156059817
// MI455X (gfx1250) — matrix operand order repaired and checked by reading, not compiled, not run on hardware
//
#include <hip/hip_runtime.h>

// Causal attention head for MI455X (gfx1250, wave32, WMMA).
//   Pass 0: W{q,k,v} fp32 [E][H] -> f16 transposed [H][E] (one-time, lives in L2)
//   Pass 1: q/k/v = x @ W in f16 (fp32 accum, v_wmma_f32_16x16x32_f16);
//           q pre-scaled by H^-0.5*log2(e) (base-2 softmax);
//           v stored transposed [B][H][T] for contiguous P@V B-fragments.
//   Pass 2: flash attention, 64-key blocks, 16 query rows per wave; causal mask
//           on the diagonal block only; softmax row-sums via P @ ones WMMA.

typedef __attribute__((ext_vector_type(16))) _Float16 v16h;
typedef __attribute__((ext_vector_type(8)))  _Float16 v8h;
typedef __attribute__((ext_vector_type(4)))  _Float16 v4h;
typedef __attribute__((ext_vector_type(8)))  float    v8f;
typedef __attribute__((ext_vector_type(4)))  float    v4f;

#define B_ 8
#define T_ 2048
#define E_ 1024
#define H_ 64
// folded into q at projection time: scores come out already in base-2 domain
#define SCL2_ (0.125f * 1.44269504088896340736f)
#define MASK_NEG (-1.0e30f)

// ---- WMMA fragment layouts (wave32, 16x16x32 f16) --------------------------
// A (16x32): lane L holds row M = L%16; with g = L/16 its 16 halves cover
//   K = 16g+e, e=0..15 (same contiguous order as B, not the ISA table order)
//   -> two contiguous 16B runs in row-major memory.
// B (32x16): lane L holds col N = L%16; K = 16*(L/16)+e, e=0..15
//   -> one contiguous 32B run per lane when K is the contiguous axis.
// C/D (16x16 f32): element r in lane L is (row = 8*(L/16)+r, col = L%16).

__device__ __forceinline__ v16h cat8(v8h lo, v8h hi) {
  v16h r;
#pragma unroll
  for (int i = 0; i < 8; ++i) { r[i] = lo[i]; r[i + 8] = hi[i]; }
  return r;
}

__device__ __forceinline__ v16h ldb32(const _Float16* p) {  // 32B B-fragment
  return cat8(*(const v8h*)p, *(const v8h*)(p + 8));
}

__device__ __forceinline__ v8f wmma_f16(v16h a, v16h b, v8f c) {
  return __builtin_amdgcn_wmma_f32_16x16x32_f16(false, a, false, b, (short)0, c,
                                                false, false);
}

// ============================================================================
// Pass 0: weight convert+transpose. fp32 [E][H] -> f16 [H][E].
// ============================================================================
__global__ __launch_bounds__(256) void wcvt(
    const float* __restrict__ wq, const float* __restrict__ wk,
    const float* __restrict__ wv, _Float16* __restrict__ tq,
    _Float16* __restrict__ tk, _Float16* __restrict__ tv) {
  const float* s = (blockIdx.y == 0) ? wq : (blockIdx.y == 1) ? wk : wv;
  _Float16*    d = (blockIdx.y == 0) ? tq : (blockIdx.y == 1) ? tk : tv;
  const int idx = blockIdx.x * 256 + threadIdx.x;   // e*H + h (coalesced read)
  d[(size_t)(idx & (H_ - 1)) * E_ + (idx >> 6)] = (_Float16)s[idx];
}

// ============================================================================
// Pass 1: q/k/v projection. grid = B*T/16 blocks x 128 threads (4 waves).
// ============================================================================
__global__ __launch_bounds__(128) void qkv_proj(
    const float* __restrict__ x, const _Float16* __restrict__ wtq,
    const _Float16* __restrict__ wtk, const _Float16* __restrict__ wtv,
    _Float16* __restrict__ qh, _Float16* __restrict__ kh,
    _Float16* __restrict__ vt) {
  constexpr int LDSS = 136;                 // halves/row: 128 + 8 pad (16B OK)
  __shared__ _Float16 xs[16 * LDSS];        // x tile staged as f16

  const int tid  = threadIdx.x;
  const int wave = tid >> 5;
  const int lane = tid & 31;
  const int g    = lane >> 4;
  const int ln   = lane & 15;
  const int row0 = blockIdx.x * 16;         // flattened (B*T) row base
  const int n0   = wave * 16;               // N tile inside H

  v8f accq = {}, acck = {}, accv = {};

  for (int k0 = 0; k0 < E_; k0 += 128) {
    __syncthreads();
    if (k0 + 128 < E_)                      // overlap next chunk with WMMA
      __builtin_prefetch(x + (size_t)(row0 + (tid >> 3)) * E_ + k0 + 128, 0, 3);
#pragma unroll
    for (int i = 0; i < 4; ++i) {
      const int idx = tid + i * 128;        // 512 float4 slots
      const int r   = idx >> 5;
      const int c4  = idx & 31;
      const v4f f = *(const v4f*)(x + (size_t)(row0 + r) * E_ + k0 + c4 * 4);
      v4h h;
#pragma unroll
      for (int j = 0; j < 4; ++j) h[j] = (_Float16)f[j];
      *(v4h*)(&xs[r * LDSS + c4 * 4]) = h;
    }
    __syncthreads();

#pragma unroll
    for (int ks = 0; ks < 4; ++ks) {
      const int kb = ks * 32;
      const _Float16* ap = &xs[ln * LDSS + kb];
      const v16h a = cat8(*(const v8h*)(ap + 16 * g),
                          *(const v8h*)(ap + 8 + 16 * g));
      const size_t woff = (size_t)(n0 + ln) * E_ + k0 + kb + (g << 4);
      accq = wmma_f16(a, ldb32(wtq + woff), accq);
      acck = wmma_f16(a, ldb32(wtk + woff), acck);
      accv = wmma_f16(a, ldb32(wtv + woff), accv);
    }
  }

  // q (pre-scaled), k: row-major [B*T][H]
#pragma unroll
  for (int r = 0; r < 8; ++r) {
    const size_t off = (size_t)(row0 + 8 * g + r) * H_ + n0 + ln;
    qh[off] = (_Float16)(accq[r] * SCL2_);
    kh[off] = (_Float16)acck[r];
  }
  // v transposed [B][H][T]: C-frag rows are consecutive t -> one 16B store
  {
    const int bb   = row0 / T_;
    const int tloc = (row0 % T_) + 8 * g;
    v8h vv;
#pragma unroll
    for (int r = 0; r < 8; ++r) vv[r] = (_Float16)accv[r];
    *(v8h*)(vt + ((size_t)bb * H_ + n0 + ln) * T_ + tloc) = vv;
  }
}

// ============================================================================
// Pass 2: causal flash attention. grid = B*T/64 blocks x 128 threads.
// Wave w owns query rows [tq0, tq0+16); iterates 64-key blocks; last block is
// the only one that applies the causal mask. Softmax denominators come from
// P @ ones via the matrix pipe (co-executes with VALU exp/rescale work).
// ============================================================================
__global__ __launch_bounds__(128) void attn(
    const _Float16* __restrict__ qh, const _Float16* __restrict__ kh,
    const _Float16* __restrict__ vt, float* __restrict__ out) {
  __shared__ _Float16 plds[4][16 * 64];     // per-wave P staging (C->A reshape)

  const int tid  = threadIdx.x;
  const int wave = tid >> 5;
  const int lane = tid & 31;
  const int g    = lane >> 4;
  const int ln   = lane & 15;

  const int tiles = T_ / 64;
  const int b     = blockIdx.x / tiles;
  const int tq0   = (blockIdx.x % tiles) * 64 + wave * 16;  // local query row
  const size_t rowbase = (size_t)b * T_;
  const _Float16* vbat = vt + (size_t)b * H_ * T_;

  // Q tile: two A fragments (head dims 0..31, 32..63): 4 x 16B loads/lane.
  v16h qa0, qa1;
  {
    const _Float16* qrow = qh + (rowbase + tq0 + ln) * H_;
    qa0 = cat8(*(const v8h*)(qrow + 16 * g), *(const v8h*)(qrow + 8 + 16 * g));
    qa1 = cat8(*(const v8h*)(qrow + 32 + 16 * g),
               *(const v8h*)(qrow + 40 + 16 * g));
  }

  v16h ones;                                // B-frag of all 1.0 for row sums
#pragma unroll
  for (int i = 0; i < 16; ++i) ones[i] = (_Float16)1.0f;

  float m[8], l[8];
  v8f o[4] = {{}, {}, {}, {}};
#pragma unroll
  for (int r = 0; r < 8; ++r) { m[r] = MASK_NEG; l[r] = 0.f; }

  _Float16* pw = &plds[wave][0];
  const int ndiag = tq0 >> 6;               // # of fully-unmasked 64-key blocks

  for (int i = 0; i <= ndiag; ++i) {
    const int s0 = i * 64;
    if (i < ndiag) {                        // prefetch next key block
      __builtin_prefetch(kh + (rowbase + s0 + 64 + lane) * H_, 0, 3);
      __builtin_prefetch(kh + (rowbase + s0 + 96 + lane) * H_, 0, 3);
      __builtin_prefetch(vbat + (size_t)(lane * 2) * T_ + s0 + 64, 0, 3);
    }

    // ---- scores (already in base-2 domain; scale folded into q) ----
    v8f s[4];
#pragma unroll
    for (int j = 0; j < 4; ++j) {
      const _Float16* krow = kh + (rowbase + s0 + 16 * j + ln) * H_;
      v8f sc = {};
      sc = wmma_f16(qa0, ldb32(krow + 16 * g), sc);
      sc = wmma_f16(qa1, ldb32(krow + 32 + 16 * g), sc);
      s[j] = sc;
    }
    if (i == ndiag) {                       // causal mask: diagonal block only
#pragma unroll
      for (int j = 0; j < 4; ++j)
#pragma unroll
        for (int r = 0; r < 8; ++r) {
          if (s0 + 16 * j + ln > tq0 + 8 * g + r) s[j][r] = MASK_NEG;
        }
    }

    // ---- row max (shuffle reduce across 16-lane groups) + rescale factor ----
    float alpha[8];
#pragma unroll
    for (int r = 0; r < 8; ++r) {
      float rm = fmaxf(fmaxf(s[0][r], s[1][r]), fmaxf(s[2][r], s[3][r]));
#pragma unroll
      for (int msk = 1; msk < 16; msk <<= 1)
        rm = fmaxf(rm, __shfl_xor(rm, msk, 32));
      const float mn = fmaxf(m[r], rm);
      alpha[r] = exp2f(m[r] - mn);
      m[r] = mn;
    }

    // ---- P = exp2(S - m); stage into LDS (C-layout -> A-layout) ----
#pragma unroll
    for (int r = 0; r < 8; ++r) {
      _Float16* prow = pw + (8 * g + r) * 64;
      prow[ln]      = (_Float16)exp2f(s[0][r] - m[r]);
      prow[16 + ln] = (_Float16)exp2f(s[1][r] - m[r]);
      prow[32 + ln] = (_Float16)exp2f(s[2][r] - m[r]);
      prow[48 + ln] = (_Float16)exp2f(s[3][r] - m[r]);
    }
    asm volatile("s_wait_dscnt 0x0" ::: "memory");  // DS in-order per wave
    const _Float16* pr = pw + ln * 64;
    const v16h pa0 = cat8(*(const v8h*)(pr + 16 * g),
                          *(const v8h*)(pr + 8 + 16 * g));
    const v16h pa1 = cat8(*(const v8h*)(pr + 32 + 16 * g),
                          *(const v8h*)(pr + 40 + 16 * g));
    asm volatile("" ::: "memory");  // next iter's stores stay after these loads

    // ---- row sums on the matrix pipe: every lane's elem r = sum(P[8g+r]) ----
    v8f rsum = {};
    rsum = wmma_f16(pa0, ones, rsum);
    rsum = wmma_f16(pa1, ones, rsum);

    // ---- bookkeeping + O rescale ----
#pragma unroll
    for (int r = 0; r < 8; ++r) {
      l[r] = l[r] * alpha[r] + rsum[r];
      o[0][r] *= alpha[r];
      o[1][r] *= alpha[r];
      o[2][r] *= alpha[r];
      o[3][r] *= alpha[r];
    }

    // ---- O += P @ V: B-frags contiguous in t thanks to transposed V ----
#pragma unroll
    for (int nt = 0; nt < 4; ++nt) {
      const _Float16* vcol = vbat + (size_t)(nt * 16 + ln) * T_ + s0;
      o[nt] = wmma_f16(pa0, ldb32(vcol + 16 * g), o[nt]);
      o[nt] = wmma_f16(pa1, ldb32(vcol + 32 + 16 * g), o[nt]);
    }
  }

  // ---- epilogue: normalize (8 divides) and store fp32 output ----
  float linv[8];
#pragma unroll
  for (int r = 0; r < 8; ++r) linv[r] = 1.0f / l[r];
#pragma unroll
  for (int nt = 0; nt < 4; ++nt) {
#pragma unroll
    for (int r = 0; r < 8; ++r) {
      out[(rowbase + tq0 + 8 * g + r) * H_ + nt * 16 + ln] =
          o[nt][r] * linv[r];
    }
  }
}

// ============================================================================
extern "C" void kernel_launch(void* const* d_in, const int* in_sizes, int n_in,
                              void* d_out, int out_size, void* d_ws,
                              size_t ws_size, hipStream_t stream) {
  const float* x  = (const float*)d_in[0];
  const float* Wk = (const float*)d_in[1];
  const float* Wq = (const float*)d_in[2];
  const float* Wv = (const float*)d_in[3];

  const size_t n  = (size_t)B_ * T_ * H_;   // 1,048,576 elems per tensor
  const size_t wn = (size_t)E_ * H_;        // 65,536 elems per weight
  _Float16* qh  = (_Float16*)d_ws;          // 2 MB (pre-scaled by SCL2_)
  _Float16* kh  = qh + n;                   // 2 MB
  _Float16* vt  = kh + n;                   // 2 MB (transposed [B][H][T])
  _Float16* wtq = vt + n;                   // 128 KB each, f16 [H][E]
  _Float16* wtk = wtq + wn;
  _Float16* wtv = wtk + wn;

  wcvt<<<dim3((E_ * H_) / 256, 3), 256, 0, stream>>>(Wq, Wk, Wv, wtq, wtk, wtv);
  qkv_proj<<<(B_ * T_) / 16, 128, 0, stream>>>(x, wtq, wtk, wtv, qh, kh, vt);
  attn<<<(B_ * T_) / 64, 128, 0, stream>>>(qh, kh, vt, (float*)d_out);
}
